// TreeLSTMAutoEncoder_32633161515610
// MI455X (gfx1250) — matrix operand order repaired and checked by reading, not compiled, not run on hardware
//
#include <hip/hip_runtime.h>
#include <hip/hip_bf16.h>
#include <math.h>

// Problem constants (from the reference)
constexpr int Bc = 4, Lc = 512, Dc = 512, Mc = 512, Vc = 32000, BRc = 3, Cc = 3;
constexpr int R2  = Bc * Lc;   // 2048 token rows
constexpr int IOU = 3 * Mc;    // 1536

typedef _Float16 h8  __attribute__((ext_vector_type(8)));
typedef _Float16 h16 __attribute__((ext_vector_type(16)));
typedef float    f8  __attribute__((ext_vector_type(8)));
typedef unsigned int u32x4 __attribute__((ext_vector_type(4)));
typedef int i32x4 __attribute__((ext_vector_type(4)));
typedef int i32x8 __attribute__((ext_vector_type(8)));

// ---------------------------------------------------------------------------
// WMMA helpers (wave32, 16x16x32 f16 -> f32). Fragment layouts per CDNA5 ISA:
//  A (16x32, row-major M x K):  lane = lh*16 + r16 holds row r16;
//      v16h elems 0..7  = K [lh*8,   lh*8+8)
//      v16h elems 8..15 = K [16+lh*8, 16+lh*8+8)
//  B (32x16, from a row-major N x K weight matrix): lane holds column r16;
//      v16h elems 0..7  = K [lh*8,   lh*8+8)
//      v16h elems 8..15 = K [16+lh*8, 16+lh*8+8)
//  D (16x16 f32): lane l, vgpr r -> row (r + 8*(l>>4)), col (l & 15)
// ---------------------------------------------------------------------------
__device__ __forceinline__ h16 load_a_frag(const _Float16* rowp, int k, int lh) {
  h8 lo = *(const h8*)(rowp + k + lh * 8);
  h8 hi = *(const h8*)(rowp + k + 16 + lh * 8);
  h16 a;
#pragma unroll
  for (int i = 0; i < 8; ++i) { a[i] = lo[i]; a[i + 8] = hi[i]; }
  return a;
}

__device__ __forceinline__ h16 load_b_frag(const _Float16* rowp, int k, int lh) {
  h8 lo = *(const h8*)(rowp + k + lh * 8);
  h8 hi = *(const h8*)(rowp + k + 16 + lh * 8);
  h16 b;
#pragma unroll
  for (int i = 0; i < 8; ++i) { b[i] = lo[i]; b[i + 8] = hi[i]; }
  return b;
}

__device__ __forceinline__ f8 wmma16(h16 a, h16 b, f8 c) {
  return __builtin_amdgcn_wmma_f32_16x16x32_f16(false, a, false, b, (short)0, c,
                                                false, false);
}

__device__ __forceinline__ float sigmf(float x) { return 1.0f / (1.0f + expf(-x)); }

// ---------------------------------------------------------------------------
// Tensor Data Mover: 2D tile load global -> LDS.
// D# per CDNA5 ISA:
//  group0: [1:0]=count=1, [63:32]=lds_addr, [120:64]=global byte addr,
//          [127:126]=type=2
//  group1: [17:16]=data_size (1 => 2 bytes), [79:48]=tensor_dim0,
//          [111:80]=tensor_dim1, [127:112]=tile_dim0, [143:128]=tile_dim1,
//          [207:160]=tensor_dim0_stride (elements)
//  groups 2/3 zero (2D tensor), cpol 0. Tracked by TENSORcnt.
// ---------------------------------------------------------------------------
__device__ __forceinline__ void tdm_load_2d(unsigned lds_off, const void* gptr,
                                            unsigned tile_d0, unsigned tile_d1,
                                            unsigned stride0_elems) {
  const unsigned td0 = 1u << 20, td1 = 1u << 20;  // large: never OOB-clip
  unsigned long long ga = (unsigned long long)gptr;
  u32x4 g0;
  g0[0] = 1u;                                        // count=1 (valid, user)
  g0[1] = lds_off;                                   // lds_addr (bytes)
  g0[2] = (unsigned)(ga & 0xFFFFFFFFu);              // global_addr[31:0]
  g0[3] = (unsigned)((ga >> 32) & 0x1FFFFFFu) | (2u << 30);  // [56:32] | type=2
  i32x8 g1;
  g1[0] = (int)(1u << 16);                           // data_size = 2 bytes
  g1[1] = (int)((td0 & 0xFFFFu) << 16);              // tensor_dim0 lo
  g1[2] = (int)((td0 >> 16) | ((td1 & 0xFFFFu) << 16));
  g1[3] = (int)((td1 >> 16) | (tile_d0 << 16));      // tile_dim0
  g1[4] = (int)(tile_d1 & 0xFFFFu);                  // tile_dim1 (tile_dim2=0)
  g1[5] = (int)stride0_elems;                        // dim0 stride [31:0]
  g1[6] = 0;                                         // stride hi, dim1 stride lo
  g1[7] = 0;
  i32x4 z = {};
#if __clang_major__ >= 23
  i32x8 z8 = {};
  __builtin_amdgcn_tensor_load_to_lds(g0, g1, z, z, z8, 0);
#else
  __builtin_amdgcn_tensor_load_to_lds(g0, g1, z, z, 0);
#endif
}

// ---------------------------------------------------------------------------
// Generic WMMA GEMM (small/medium): C[r][n] = sum_k A[r][k]*Bm[n][k] (+bias)
// grid = (N/16, R/16), block = 32 (one wave per 16x16 tile).
// ---------------------------------------------------------------------------
__global__ __launch_bounds__(32) void k_gemm_wmma(
    const _Float16* __restrict__ A, const _Float16* __restrict__ Bm,
    const float* __restrict__ bias, float* __restrict__ C, int K, int ldc) {
  const int lane = threadIdx.x & 31;
  const int lh = lane >> 4, r16 = lane & 15;
  const size_t mbase = (size_t)blockIdx.y * 16;
  const size_t nbase = (size_t)blockIdx.x * 16;
  const _Float16* arow = A + (mbase + r16) * (size_t)K;
  const _Float16* brow = Bm + (nbase + r16) * (size_t)K;
  f8 acc = {};
  for (int k = 0; k < K; k += 32) {
    __builtin_prefetch(arow + k + 128, 0, 1);   // global_prefetch_b8
    __builtin_prefetch(brow + k + 128, 0, 1);
    h16 af = load_a_frag(arow, k, lh);
    h16 bf = load_b_frag(brow, k, lh);
    acc = wmma16(af, bf, acc);
  }
  const float bv = bias ? bias[nbase + r16] : 0.0f;
  float* crow = C + (mbase + 8 * lh) * (size_t)ldc + nbase + r16;
#pragma unroll
  for (int r = 0; r < 8; ++r) crow[(size_t)r * ldc] = acc[r] + bv;
}

// ---------------------------------------------------------------------------
// TDM-staged block GEMM for the dominant logits matmul.
// Block tile 128(M) x 128(N), K = 512 staged in 4 chunks of 128 via the
// Tensor Data Mover into double-buffered LDS (4 x 32 KB). 256 threads =
// 8 waves in a 4(M) x 2(N) grid; each wave owns a 32x64 sub-tile
// (2x4 WMMA accumulators -> 4x A reuse, 2x B reuse from LDS).
// grid = (N/128, R/128). Dynamic LDS = 131072 B.
// ---------------------------------------------------------------------------
__global__ __launch_bounds__(256) void k_gemm_tdm(
    const _Float16* __restrict__ A, const _Float16* __restrict__ Bm,
    float* __restrict__ C, int ldc) {
  extern __shared__ char smem[];
  _Float16* Ab0 = (_Float16*)(smem);            // 32768 B
  _Float16* Ab1 = (_Float16*)(smem + 32768);    // 32768 B
  _Float16* Bb0 = (_Float16*)(smem + 65536);    // 32768 B
  _Float16* Bb1 = (_Float16*)(smem + 98304);    // 32768 B

  const int tid = threadIdx.x;
  const int wv = tid >> 5, lane = tid & 31;
  const int lh = lane >> 4, r16 = lane & 15;
  const int wm = wv >> 1, wn = wv & 1;          // 4 x 2 wave grid
  const size_t mblk = (size_t)blockIdx.y * 128;
  const size_t nblk = (size_t)blockIdx.x * 128;
  const _Float16* Ablk = A + mblk * (size_t)Dc;
  const _Float16* Bblk = Bm + nblk * (size_t)Dc;

  f8 acc00 = {}, acc01 = {}, acc02 = {}, acc03 = {};
  f8 acc10 = {}, acc11 = {}, acc12 = {}, acc13 = {};

  // prologue: stage chunk 0 into buffer 0
  if (wv == 0) {
    tdm_load_2d(0u,      Ablk, 128u, 128u, (unsigned)Dc);
    tdm_load_2d(65536u,  Bblk, 128u, 128u, (unsigned)Dc);
  }

#pragma unroll
  for (int kc = 0; kc < 4; ++kc) {
    const int cur = kc & 1;
    if (wv == 0) __builtin_amdgcn_s_wait_tensorcnt(0);  // current buffers ready
    __syncthreads();
    if (kc < 3 && wv == 0) {  // stage next chunk into the other buffer
      unsigned ao = (cur == 0) ? 32768u : 0u;
      unsigned bo = (cur == 0) ? 98304u : 65536u;
      tdm_load_2d(ao, Ablk + (kc + 1) * 128, 128u, 128u, (unsigned)Dc);
      tdm_load_2d(bo, Bblk + (kc + 1) * 128, 128u, 128u, (unsigned)Dc);
    }
    const _Float16* As = (cur == 0) ? Ab0 : Ab1;
    const _Float16* Bs = (cur == 0) ? Bb0 : Bb1;
#pragma unroll
    for (int kk = 0; kk < 128; kk += 32) {
      h16 a0 = load_a_frag(As + (wm * 32 + 0 + r16) * 128, kk, lh);
      h16 a1 = load_a_frag(As + (wm * 32 + 16 + r16) * 128, kk, lh);
      h16 b0 = load_b_frag(Bs + (wn * 64 + 0 + r16) * 128, kk, lh);
      h16 b1 = load_b_frag(Bs + (wn * 64 + 16 + r16) * 128, kk, lh);
      h16 b2 = load_b_frag(Bs + (wn * 64 + 32 + r16) * 128, kk, lh);
      h16 b3 = load_b_frag(Bs + (wn * 64 + 48 + r16) * 128, kk, lh);
      acc00 = wmma16(a0, b0, acc00);
      acc01 = wmma16(a0, b1, acc01);
      acc02 = wmma16(a0, b2, acc02);
      acc03 = wmma16(a0, b3, acc03);
      acc10 = wmma16(a1, b0, acc10);
      acc11 = wmma16(a1, b1, acc11);
      acc12 = wmma16(a1, b2, acc12);
      acc13 = wmma16(a1, b3, acc13);
    }
    __syncthreads();  // all reads of `cur` done before it is restaged
  }

  // epilogue: D row = mt*16 + r + 8*lh, col = nt*16 + r16
  const size_t cbase = (mblk + wm * 32 + 8 * lh) * (size_t)ldc + nblk + wn * 64 + r16;
  f8* accs[8] = {&acc00, &acc01, &acc02, &acc03, &acc10, &acc11, &acc12, &acc13};
#pragma unroll
  for (int mt = 0; mt < 2; ++mt) {
#pragma unroll
    for (int nt = 0; nt < 4; ++nt) {
      f8 v = *accs[mt * 4 + nt];
      float* crow = C + cbase + (size_t)mt * 16 * ldc + nt * 16;
#pragma unroll
      for (int r = 0; r < 8; ++r) crow[(size_t)r * ldc] = v[r];
    }
  }
}

// ---------------------------------------------------------------------------
// Elementwise prep kernels
// ---------------------------------------------------------------------------
__global__ void k_cvt_f16(_Float16* dst, const float* src, size_t n) {
  size_t i = (size_t)blockIdx.x * 256 + threadIdx.x;
  if (i < n) dst[i] = (_Float16)src[i];
}

// src shape (O, Md, BR) -> dst[br][o][m]
__global__ void k_cvt_branch(_Float16* dst, const float* src, int O, int Md) {
  size_t n = (size_t)BRc * O * Md;
  size_t i = (size_t)blockIdx.x * 256 + threadIdx.x;
  if (i >= n) return;
  size_t per = (size_t)O * Md;
  int br = (int)(i / per);
  size_t rem = i - (size_t)br * per;  // o*Md + m
  dst[i] = (_Float16)src[rem * BRc + br];
}

__global__ void k_add_bias(float* dst, const float* a, const float* b, int n) {
  int i = blockIdx.x * 256 + threadIdx.x;
  if (i < n) dst[i] = a[i] + b[i];
}

// dst[t*Dc + d] = f16(emb[idx[t]*Dc + d]) over n = ntok*Dc
__global__ void k_gather_emb(_Float16* dst, const float* __restrict__ emb,
                             const int* __restrict__ idx, size_t n) {
  size_t i = (size_t)blockIdx.x * 256 + threadIdx.x;
  if (i >= n) return;
  size_t t = i / Dc, d = i - t * Dc;
  dst[i] = (_Float16)emb[(size_t)idx[t] * Dc + d];
}

// h_dec f16 extraction: dst[t*Mc+m] = dec_out[t*2M + M + m]
__global__ void k_extract_hdec(_Float16* dst, const float* __restrict__ dec_out,
                               size_t n) {
  size_t i = (size_t)blockIdx.x * 256 + threadIdx.x;
  if (i >= n) return;
  size_t t = i / Mc, m = i - t * Mc;
  dst[i] = (_Float16)dec_out[t * (size_t)(2 * Mc) + Mc + m];
}

// ---------------------------------------------------------------------------
// Encoder scan: one workgroup, 1024 threads = 32 waves, sequential over steps.
// ---------------------------------------------------------------------------
__global__ __launch_bounds__(1024) void k_enc_scan(
    const int* __restrict__ x_c, const float* __restrict__ x_m,
    const float* __restrict__ x_m_c, const float* __restrict__ pre_iou,
    const float* __restrict__ pre_f, const _Float16* __restrict__ Wiouh,
    const _Float16* __restrict__ Wfh, float* __restrict__ enc_out,
    const float* __restrict__ disc_W, const float* __restrict__ disc_b,
    const float* __restrict__ init_W, const float* __restrict__ init_b,
    float* __restrict__ d_tail, float* __restrict__ dec_out) {
  extern __shared__ char smem[];
  _Float16* S  = (_Float16*)(smem);             // 16384 B
  _Float16* Ch = (_Float16*)(smem + 16384);     // 16384 B
  float* chf  = (float*)(smem + 32768);         // 24576 B
  float* cs   = (float*)(smem + 57344);         // 24576 B
  float* iouR = (float*)(smem + 81920);         // 24576 B
  float* fR   = (float*)(smem + 106496);        // 24576 B
  float* prev = (float*)(smem + 131072);        // 16384 B

  const int tid = threadIdx.x;
  const int wv = tid >> 5, lane = tid & 31;
  const int lh = lane >> 4, r16 = lane & 15;

  for (int i = tid; i < 16 * Mc; i += 1024) { S[i] = (_Float16)0.f; Ch[i] = (_Float16)0.f; }
  for (int i = tid; i < 4 * 2 * Mc; i += 1024) prev[i] = 0.f;
  __syncthreads();

  for (int step = 1; step < Lc; ++step) {
    // ---- gather children ----
    for (int w = tid; w < 12 * Mc; w += 1024) {
      int row = w >> 9, m = w & 511;     // row = b*3 + c
      int b = row / Cc, c = row - b * Cc;
      int idx = x_c[(b * Lc + step) * Cc + c];
      const float* src = enc_out + ((size_t)(b * Lc + idx)) * (2 * Mc);
      float csv = src[m], chv = src[Mc + m];
      cs[row * Mc + m] = csv;
      chf[row * Mc + m] = chv;
      Ch[row * Mc + m] = (_Float16)chv;
    }
    __syncthreads();
    for (int w = tid; w < 4 * Mc; w += 1024) {
      int b = w >> 9, m = w & 511;
      const float* xmc = x_m_c + (b * Lc + step) * Cc;
      float s = chf[(b * 3 + 0) * Mc + m] * xmc[0] +
                chf[(b * 3 + 1) * Mc + m] * xmc[1] +
                chf[(b * 3 + 2) * Mc + m] * xmc[2];
      S[b * Mc + m] = (_Float16)s;
    }
    __syncthreads();

    // ---- recurrent matmuls via WMMA: 96 iou tiles + 32 f tiles = 128 ----
#pragma unroll
    for (int ti = 0; ti < 4; ++ti) {
      int t = wv * 4 + ti;
      const bool is_iou = (t < 96);
      const _Float16* arow = (is_iou ? S : Ch) + r16 * Mc;
      const _Float16* brow =
          is_iou ? (Wiouh + ((size_t)(t * 16 + r16)) * Mc)
                 : (Wfh + ((size_t)((t - 96) * 16 + r16)) * Mc);
      f8 acc = {};
      for (int k = 0; k < Mc; k += 32)
        acc = wmma16(load_a_frag(arow, k, lh), load_b_frag(brow, k, lh), acc);
      if (is_iou) {
        if (lh == 0) {
#pragma unroll
          for (int r = 0; r < 4; ++r) iouR[r * IOU + t * 16 + r16] = acc[r];
        }
      } else {
        int ft = t - 96;
        if (lh == 0) {
#pragma unroll
          for (int r = 0; r < 8; ++r) fR[r * Mc + ft * 16 + r16] = acc[r];
        } else {
#pragma unroll
          for (int r = 0; r < 4; ++r) fR[(8 + r) * Mc + ft * 16 + r16] = acc[r];
        }
      }
    }
    __syncthreads();

    // ---- epilogue: gates, cell, mask blend ----
    for (int w = tid; w < 4 * Mc; w += 1024) {
      int b = w >> 9, m = w & 511;
      const float* pio = pre_iou + ((size_t)(b * Lc + step)) * IOU;
      float ig = sigmf(pio[m] + iouR[b * IOU + m]);
      float og = sigmf(pio[Mc + m] + iouR[b * IOU + Mc + m]);
      float ug = tanhf(pio[2 * Mc + m] + iouR[b * IOU + 2 * Mc + m]);
      float pf = pre_f[((size_t)(b * Lc + step)) * Mc + m];
      const float* xmc = x_m_c + (b * Lc + step) * Cc;
      float cacc = ig * ug;
#pragma unroll
      for (int c = 0; c < Cc; ++c) {
        float fg = sigmf(fR[(b * 3 + c) * Mc + m] + pf);
        cacc += fg * cs[(b * 3 + c) * Mc + m] * xmc[c];
      }
      float hv = og * tanhf(cacc);
      float xm = x_m[b * Lc + step];
      float sc = xm * cacc + (1.f - xm) * prev[b * 2 * Mc + m];
      float sh = xm * hv + (1.f - xm) * prev[b * 2 * Mc + Mc + m];
      prev[b * 2 * Mc + m] = sc;
      prev[b * 2 * Mc + Mc + m] = sh;
      float* dst = enc_out + ((size_t)(b * Lc + step)) * (2 * Mc);
      dst[m] = sc;
      dst[Mc + m] = sh;
    }
    __threadfence();
    __syncthreads();
  }

  // ---- discriminator + init projections (tiny, VALU) ----
  float* trep = iouR;  // reuse LDS (needs 2048 f32)
  for (int w = tid; w < 4 * Mc; w += 1024) {
    int b = w >> 9, j = w & 511;
    const float* hrow = prev + b * 2 * Mc + Mc;  // tree_encoding = final h
    const float* wrow = disc_W + (size_t)j * Mc;
    float acc = disc_b[j];
    for (int m = 0; m < Mc; ++m) acc += hrow[m] * wrow[m];
    trep[b * Mc + j] = acc;
    d_tail[4 + b * Mc + j] = acc;  // tree_repr output
  }
  if (tid < 4) d_tail[tid] = 1.0f;  // tree_codes
  __syncthreads();
  for (int w = tid; w < 4 * 2 * Mc; w += 1024) {
    int b = w >> 10, j2 = w & 1023;
    const float* wrow = init_W + (size_t)j2 * Mc;
    float acc = init_b[j2];
    for (int j = 0; j < Mc; ++j) acc += trep[b * Mc + j] * wrow[j];
    dec_out[((size_t)(b * Lc)) * (2 * Mc) + j2] = acc;  // root decoder state
  }
}

// ---------------------------------------------------------------------------
// Decoder scan: one workgroup, 1024 threads = 32 waves.
// ---------------------------------------------------------------------------
__global__ __launch_bounds__(1024) void k_dec_scan(
    const int* __restrict__ y_p, const int* __restrict__ y_r,
    const float* __restrict__ y_m, const float* __restrict__ pre_iou,
    const float* __restrict__ pre_f, const _Float16* __restrict__ wbiou,
    const _Float16* __restrict__ wbf, const float* __restrict__ bb_iou,
    const float* __restrict__ bb_f, float* __restrict__ dec_out) {
  extern __shared__ char smem[];
  _Float16* Ah = (_Float16*)(smem);           // 16384 B
  float* ps   = (float*)(smem + 16384);       // 8192 B
  float* phf  = (float*)(smem + 24576);       // 8192 B
  float* iouR = (float*)(smem + 32768);       // 73728 B
  float* fR   = (float*)(smem + 106496);      // 24576 B

  const int tid = threadIdx.x;
  const int wv = tid >> 5, lane = tid & 31;
  const int lh = lane >> 4, r16 = lane & 15;

  for (int i = tid; i < 16 * Mc; i += 1024) Ah[i] = (_Float16)0.f;
  __syncthreads();

  for (int step = 1; step < Lc; ++step) {
    // ---- gather parent state ----
    for (int w = tid; w < 4 * Mc; w += 1024) {
      int b = w >> 9, m = w & 511;
      int yp = y_p[b * Lc + step];
      const float* pr = dec_out + ((size_t)(b * Lc + yp)) * (2 * Mc);
      ps[b * Mc + m] = pr[m];
      float ph = pr[Mc + m];
      phf[b * Mc + m] = ph;
      Ah[b * Mc + m] = (_Float16)ph;
    }
    __syncthreads();

    // ---- 3 branches x (96 iou + 32 f) = 384 tiles; 12 per wave ----
#pragma unroll
    for (int ti = 0; ti < 12; ++ti) {
      int t = wv * 12 + ti;
      int br = t >> 7;       // / 128
      int tt = t & 127;
      const bool is_iou = (tt < 96);
      const _Float16* arow = Ah + r16 * Mc;
      const _Float16* brow =
          is_iou ? (wbiou + ((size_t)br * IOU + tt * 16 + r16) * Mc)
                 : (wbf + ((size_t)br * Mc + (tt - 96) * 16 + r16) * Mc);
      f8 acc = {};
      for (int k = 0; k < Mc; k += 32)
        acc = wmma16(load_a_frag(arow, k, lh), load_b_frag(brow, k, lh), acc);
      if (lh == 0) {
        if (is_iou) {
#pragma unroll
          for (int r = 0; r < 4; ++r)
            iouR[(br * 4 + r) * IOU + tt * 16 + r16] = acc[r];
        } else {
#pragma unroll
          for (int r = 0; r < 4; ++r)
            fR[(br * 4 + r) * Mc + (tt - 96) * 16 + r16] = acc[r];
        }
      }
    }
    __syncthreads();

    // ---- epilogue: per-sample branch select, gates, blend ----
    for (int w = tid; w < 4 * Mc; w += 1024) {
      int b = w >> 9, m = w & 511;
      int yp = y_p[b * Lc + step];
      int order = ((unsigned)y_r[b * Lc + step]) % BRc;
      const float* pio = pre_iou + ((size_t)(b * Lc + yp)) * IOU;
      float ig = sigmf(pio[m] + iouR[(order * 4 + b) * IOU + m] +
                       bb_iou[(size_t)m * BRc + order]);
      float og = sigmf(pio[Mc + m] + iouR[(order * 4 + b) * IOU + Mc + m] +
                       bb_iou[(size_t)(Mc + m) * BRc + order]);
      float ug = tanhf(pio[2 * Mc + m] + iouR[(order * 4 + b) * IOU + 2 * Mc + m] +
                       bb_iou[(size_t)(2 * Mc + m) * BRc + order]);
      float pf = pre_f[((size_t)(b * Lc + yp)) * Mc + m];
      float fg = sigmf(fR[(order * 4 + b) * Mc + m] +
                       bb_f[(size_t)m * BRc + order] + pf);
      float cv = ig * ug + fg * ps[b * Mc + m];
      float hv = og * tanhf(cv);
      float ym = y_m[b * Lc + step];
      float sc = ym * cv + (1.f - ym) * ps[b * Mc + m];
      float sh = ym * hv + (1.f - ym) * phf[b * Mc + m];
      float* dst = dec_out + ((size_t)(b * Lc + step)) * (2 * Mc);
      dst[m] = sc;
      dst[Mc + m] = sh;
    }
    __threadfence();
    __syncthreads();
  }
}

// ---------------------------------------------------------------------------
// Host launch
// ---------------------------------------------------------------------------
extern "C" void kernel_launch(void* const* d_in, const int* in_sizes, int n_in,
                              void* d_out, int out_size, void* d_ws, size_t ws_size,
                              hipStream_t stream) {
  (void)in_sizes; (void)n_in; (void)out_size; (void)ws_size;
  const int* x   = (const int*)d_in[0];
  const int* y   = (const int*)d_in[1];
  const int* x_c = (const int*)d_in[2];
  const int* y_p = (const int*)d_in[3];
  const int* y_r = (const int*)d_in[4];
  const float* x_m   = (const float*)d_in[5];
  const float* x_m_c = (const float*)d_in[6];
  const float* y_m   = (const float*)d_in[7];
  const float* emb   = (const float*)d_in[8];
  const float* enc_Wioux = (const float*)d_in[9];
  const float* enc_bioux = (const float*)d_in[10];
  const float* enc_Wiouh = (const float*)d_in[11];
  const float* enc_biouh = (const float*)d_in[12];
  const float* enc_Wfx = (const float*)d_in[13];
  const float* enc_bfx = (const float*)d_in[14];
  const float* enc_Wfh = (const float*)d_in[15];
  const float* enc_bfh = (const float*)d_in[16];
  const float* dec_Wioux = (const float*)d_in[17];
  const float* dec_bioux = (const float*)d_in[18];
  const float* dec_Wfx = (const float*)d_in[19];
  const float* dec_bfx = (const float*)d_in[20];
  const float* dec_Wb_iou = (const float*)d_in[21];
  const float* dec_bb_iou = (const float*)d_in[22];
  const float* dec_Wb_f = (const float*)d_in[23];
  const float* dec_bb_f = (const float*)d_in[24];
  const float* disc_W = (const float*)d_in[25];
  const float* disc_b = (const float*)d_in[26];
  const float* init_W = (const float*)d_in[27];
  const float* init_b = (const float*)d_in[28];
  float* out = (float*)d_out;

  // ---- workspace bump allocator (256B aligned) ----
  char* base = (char*)d_ws;
  size_t off = 0;
  auto alloc = [&](size_t bytes) -> void* {
    void* p = base + off;
    off = (off + bytes + 255) & ~(size_t)255;
    return p;
  };
  auto allocH = [&](size_t n) { return (_Float16*)alloc(n * sizeof(_Float16)); };
  auto allocF = [&](size_t n) { return (float*)alloc(n * sizeof(float)); };

  _Float16* xe_h     = allocH((size_t)R2 * Dc);
  _Float16* ye_h     = allocH((size_t)R2 * Dc);
  _Float16* Wioux_h  = allocH((size_t)IOU * Dc);
  _Float16* Wiouh_h  = allocH((size_t)IOU * Mc);
  _Float16* Wfx_h    = allocH((size_t)Mc * Dc);
  _Float16* Wfh_h    = allocH((size_t)Mc * Mc);
  _Float16* dWioux_h = allocH((size_t)IOU * Dc);
  _Float16* dWfx_h   = allocH((size_t)Mc * Dc);
  _Float16* wbiou_h  = allocH((size_t)BRc * IOU * Mc);
  _Float16* wbf_h    = allocH((size_t)BRc * Mc * Mc);
  _Float16* emb_h    = allocH((size_t)Vc * Dc);
  _Float16* hdec_h   = allocH((size_t)R2 * Mc);
  float* enc_pre_iou = allocF((size_t)R2 * IOU);
  float* enc_pre_f   = allocF((size_t)R2 * Mc);
  float* dec_pre_iou = allocF((size_t)R2 * IOU);
  float* dec_pre_f   = allocF((size_t)R2 * Mc);
  float* enc_out     = allocF((size_t)R2 * 2 * Mc);
  float* dec_out     = allocF((size_t)R2 * 2 * Mc);
  float* b_enc_iou   = allocF(IOU);
  float* b_enc_f     = allocF(Mc);

  auto g1 = [](size_t n) { return dim3((unsigned)((n + 255) / 256)); };

  // ---- zero the scan output buffers (reference initializes to zeros) ----
  hipMemsetAsync(enc_out, 0, (size_t)R2 * 2 * Mc * sizeof(float), stream);
  hipMemsetAsync(dec_out, 0, (size_t)R2 * 2 * Mc * sizeof(float), stream);

  // ---- precision conversion + embedding gathers ----
  k_cvt_f16<<<g1((size_t)IOU * Dc), 256, 0, stream>>>(Wioux_h, enc_Wioux, (size_t)IOU * Dc);
  k_cvt_f16<<<g1((size_t)IOU * Mc), 256, 0, stream>>>(Wiouh_h, enc_Wiouh, (size_t)IOU * Mc);
  k_cvt_f16<<<g1((size_t)Mc * Dc), 256, 0, stream>>>(Wfx_h, enc_Wfx, (size_t)Mc * Dc);
  k_cvt_f16<<<g1((size_t)Mc * Mc), 256, 0, stream>>>(Wfh_h, enc_Wfh, (size_t)Mc * Mc);
  k_cvt_f16<<<g1((size_t)IOU * Dc), 256, 0, stream>>>(dWioux_h, dec_Wioux, (size_t)IOU * Dc);
  k_cvt_f16<<<g1((size_t)Mc * Dc), 256, 0, stream>>>(dWfx_h, dec_Wfx, (size_t)Mc * Dc);
  k_cvt_f16<<<g1((size_t)Vc * Dc), 256, 0, stream>>>(emb_h, emb, (size_t)Vc * Dc);
  k_cvt_branch<<<g1((size_t)BRc * IOU * Mc), 256, 0, stream>>>(wbiou_h, dec_Wb_iou, IOU, Mc);
  k_cvt_branch<<<g1((size_t)BRc * Mc * Mc), 256, 0, stream>>>(wbf_h, dec_Wb_f, Mc, Mc);
  k_add_bias<<<g1(IOU), 256, 0, stream>>>(b_enc_iou, enc_bioux, enc_biouh, IOU);
  k_add_bias<<<g1(Mc), 256, 0, stream>>>(b_enc_f, enc_bfx, enc_bfh, Mc);
  k_gather_emb<<<g1((size_t)R2 * Dc), 256, 0, stream>>>(xe_h, emb, x, (size_t)R2 * Dc);
  k_gather_emb<<<g1((size_t)R2 * Dc), 256, 0, stream>>>(ye_h, emb, y, (size_t)R2 * Dc);

  // ---- hoisted input-side GEMMs (biases folded in) ----
  k_gemm_wmma<<<dim3(IOU / 16, R2 / 16), 32, 0, stream>>>(xe_h, Wioux_h, b_enc_iou,
                                                          enc_pre_iou, Dc, IOU);
  k_gemm_wmma<<<dim3(Mc / 16, R2 / 16), 32, 0, stream>>>(xe_h, Wfx_h, b_enc_f,
                                                         enc_pre_f, Dc, Mc);
  k_gemm_wmma<<<dim3(IOU / 16, R2 / 16), 32, 0, stream>>>(ye_h, dWioux_h, dec_bioux,
                                                          dec_pre_iou, Dc, IOU);
  k_gemm_wmma<<<dim3(Mc / 16, R2 / 16), 32, 0, stream>>>(ye_h, dWfx_h, dec_bfx,
                                                         dec_pre_f, Dc, Mc);

  // ---- sequential scans (one WGP workgroup each, big dynamic LDS) ----
  float* d_tail = out + (size_t)R2 * Vc;  // tree_codes (4) + tree_repr (2048)
  hipFuncSetAttribute((const void*)k_enc_scan,
                      hipFuncAttributeMaxDynamicSharedMemorySize, 147456);
  hipFuncSetAttribute((const void*)k_dec_scan,
                      hipFuncAttributeMaxDynamicSharedMemorySize, 131072);
  hipFuncSetAttribute((const void*)k_gemm_tdm,
                      hipFuncAttributeMaxDynamicSharedMemorySize, 131072);
  k_enc_scan<<<1, 1024, 147456, stream>>>(x_c, x_m, x_m_c, enc_pre_iou, enc_pre_f,
                                          Wiouh_h, Wfh_h, enc_out, disc_W, disc_b,
                                          init_W, init_b, d_tail, dec_out);
  k_dec_scan<<<1, 1024, 131072, stream>>>(y_p, y_r, y_m, dec_pre_iou, dec_pre_f,
                                          wbiou_h, wbf_h, dec_bb_iou, dec_bb_f,
                                          dec_out);

  // ---- logits = h_dec @ emb^T : dominant GEMM, TDM-staged block WMMA ----
  k_extract_hdec<<<g1((size_t)R2 * Mc), 256, 0, stream>>>(hdec_h, dec_out,
                                                          (size_t)R2 * Mc);
  k_gemm_tdm<<<dim3(Vc / 128, R2 / 128), 256, 131072, stream>>>(hdec_h, emb_h,
                                                                out, Vc);
}
